// ApplicationScoringLayer_61486751809628
// MI455X (gfx1250) — matrix operand order repaired and checked by reading, not compiled, not run on hardware
//
#include <hip/hip_runtime.h>
#include <hip/hip_bf16.h>

typedef __attribute__((ext_vector_type(16))) _Float16 v16h;
typedef __attribute__((ext_vector_type(8)))  _Float16 v8h;
typedef __attribute__((ext_vector_type(8)))  float    v8f;

#define NUM_R     8192
#define NUM_L     8192
#define NUM_K     64
#define VOCAB     100000

struct alignas(4) h2pack { _Float16 a, b; };

// ---------------------------------------------------------------------------
// Kernel 1: per-type logsumexp over the full 100k vocab of (s[k,:] + bias[:])
// 64 blocks x 256 threads, each block reduces one type row.
// ---------------------------------------------------------------------------
__global__ __launch_bounds__(256) void lse_kernel(const float* __restrict__ s,
                                                  const float* __restrict__ bias,
                                                  float* __restrict__ lse) {
    __shared__ float red[256];
    const int k = blockIdx.x;
    const float* row = s + (size_t)k * VOCAB;
    const int t = threadIdx.x;

    float m = -3.4e38f;
    for (int v = t; v < VOCAB; v += 256) m = fmaxf(m, row[v] + bias[v]);
    red[t] = m; __syncthreads();
    for (int o = 128; o > 0; o >>= 1) {
        if (t < o) red[t] = fmaxf(red[t], red[t + o]);
        __syncthreads();
    }
    const float bm = red[0];
    __syncthreads();

    float sum = 0.0f;
    for (int v = t; v < VOCAB; v += 256) sum += __expf(row[v] + bias[v] - bm);
    red[t] = sum; __syncthreads();
    for (int o = 128; o > 0; o >>= 1) {
        if (t < o) red[t] += red[t + o];
        __syncthreads();
    }
    if (t == 0) lse[k] = bm + __logf(red[0]);
}

// ---------------------------------------------------------------------------
// Kernel 2: A-side. One wave (32 lanes) per rhs row; lane handles k=2*lane,2*lane+1.
// expA[r][k] = exp(x - max_k x)   (row-major f16, stride 64)
// ma[r]      = max_k logsoftmax   = -log(sum exp(x - max))
// ---------------------------------------------------------------------------
__global__ __launch_bounds__(256) void prep_a_kernel(const float* __restrict__ rts,
                                                     const int* __restrict__ ridx,
                                                     _Float16* __restrict__ expA,
                                                     float* __restrict__ ma) {
    const int lane = threadIdx.x & 31;
    const int r = (blockIdx.x * 256 + threadIdx.x) >> 5;
    const int g = ridx[r];
    const float* row = rts + (size_t)g * NUM_K;
    const float x0 = row[2 * lane];
    const float x1 = row[2 * lane + 1];

    float m = fmaxf(x0, x1);
    for (int o = 16; o > 0; o >>= 1) m = fmaxf(m, __shfl_xor(m, o, 32));
    const float e0 = __expf(x0 - m);
    const float e1 = __expf(x1 - m);
    float sum = e0 + e1;
    for (int o = 16; o > 0; o >>= 1) sum += __shfl_xor(sum, o, 32);

    h2pack p; p.a = (_Float16)e0; p.b = (_Float16)e1;
    *(h2pack*)(expA + (size_t)r * NUM_K + 2 * lane) = p;
    if (lane == 0) ma[r] = -__logf(sum);
}

// ---------------------------------------------------------------------------
// Kernel 3: B-side. One wave per lhs column j; lane handles k=2*lane,2*lane+1.
// x[k] = s[k, idx] + bias[idx] - lse[k]  (this IS logsoftmax over full vocab)
// expB stored N-major: expB[j][k], stride 64. mb[j] = max_k x.
// ---------------------------------------------------------------------------
__global__ __launch_bounds__(256) void prep_b_kernel(const float* __restrict__ s,
                                                     const float* __restrict__ bias,
                                                     const int* __restrict__ lidx,
                                                     const float* __restrict__ lse,
                                                     _Float16* __restrict__ expB,
                                                     float* __restrict__ mb) {
    const int lane = threadIdx.x & 31;
    const int j = (blockIdx.x * 256 + threadIdx.x) >> 5;
    const int g = lidx[j];
    const float bv = bias[g];
    const int k0 = 2 * lane;
    const float x0 = s[(size_t)k0 * VOCAB + g]       + bv - lse[k0];
    const float x1 = s[(size_t)(k0 + 1) * VOCAB + g] + bv - lse[k0 + 1];

    float m = fmaxf(x0, x1);
    for (int o = 16; o > 0; o >>= 1) m = fmaxf(m, __shfl_xor(m, o, 32));
    const float e0 = __expf(x0 - m);
    const float e1 = __expf(x1 - m);

    h2pack p; p.a = (_Float16)e0; p.b = (_Float16)e1;
    *(h2pack*)(expB + (size_t)j * NUM_K + 2 * lane) = p;
    if (lane == 0) mb[j] = m;
}

// ---------------------------------------------------------------------------
// Kernel 4: WMMA GEMM + log epilogue, transposed store.
// One 16x16 tile of C per wave; K=64 via two v_wmma_f32_16x16x32_f16.
// out[(l0+n)*8192 + r0+m] = log(acc[m][n]) + ma[r0+m] + mb[l0+n]
// acc is a sum of 64 positive products with max term ~= 1 (both operands are
// exp(logp - max) in (0,1], and each row/col contains a 1.0), so acc is in
// ~[0.5, 64]: no denorm / inf / zero cases -> raw v_log_f32 (log2) is safe.
// ---------------------------------------------------------------------------
__global__ __launch_bounds__(256) void gemm_log_kernel(const _Float16* __restrict__ expA,
                                                       const _Float16* __restrict__ expB,
                                                       const float* __restrict__ ma,
                                                       const float* __restrict__ mb,
                                                       float* __restrict__ out) {
    const int lane = threadIdx.x & 31;
    const int wave = threadIdx.x >> 5;
    const int tile = blockIdx.x * 8 + wave;
    const int tm = tile & 511;         // M tile index (r)
    const int tn = tile >> 9;          // N tile index (l)
    const int r0 = tm * 16;
    const int l0 = tn * 16;

    const int mn = lane & 15;          // A: row m; B: col n; same lane sub-index
    const int hi = lane >> 4;

    // A fragment (16x32 f16): lane row = r0+mn; 16 contiguous K at 16*hi
    const _Float16* pA = expA + (size_t)(r0 + mn) * NUM_K;
    union UA { v16h v; v8h h[2]; };
    UA a0, a1;
    a0.h[0] = *(const v8h*)(pA + 16 * hi);
    a0.h[1] = *(const v8h*)(pA + 8 + 16 * hi);
    a1.h[0] = *(const v8h*)(pA + 32 + 16 * hi);
    a1.h[1] = *(const v8h*)(pA + 40 + 16 * hi);

    // B fragment (32x16 f16): lane col = l0+mn; 16 contiguous K at 16*hi
    const _Float16* pB = expB + (size_t)(l0 + mn) * NUM_K;
    v16h b0 = *(const v16h*)(pB + 16 * hi);
    v16h b1 = *(const v16h*)(pB + 32 + 16 * hi);

    v8f acc = {};
    acc = __builtin_amdgcn_wmma_f32_16x16x32_f16(false, a0.v, false, b0,
                                                 (short)0, acc, false, false);
    acc = __builtin_amdgcn_wmma_f32_16x16x32_f16(false, a1.v, false, b1,
                                                 (short)0, acc, false, false);

    // Epilogue: acc[i] = C[r0 + 8*hi + i][l0 + mn]
    const int rbase = r0 + 8 * hi;
    const float4 ma0 = *(const float4*)(ma + rbase);
    const float4 ma1 = *(const float4*)(ma + rbase + 4);
    const float mbv = mb[l0 + mn];

    constexpr float LN2 = 0.69314718055994530942f;
    // ma + mb folded once; log(x) = log2(x) * ln2 via raw v_log_f32 + one FMA
    float add[8];
    add[0] = ma0.x + mbv; add[1] = ma0.y + mbv;
    add[2] = ma0.z + mbv; add[3] = ma0.w + mbv;
    add[4] = ma1.x + mbv; add[5] = ma1.y + mbv;
    add[6] = ma1.z + mbv; add[7] = ma1.w + mbv;

    float o[8];
#pragma unroll
    for (int i = 0; i < 8; ++i)
        o[i] = fmaf(__builtin_amdgcn_logf(acc[i]), LN2, add[i]);

    float4 o0, o1;
    o0.x = o[0]; o0.y = o[1]; o0.z = o[2]; o0.w = o[3];
    o1.x = o[4]; o1.y = o[5]; o1.z = o[6]; o1.w = o[7];

    float* po = out + (size_t)(l0 + mn) * NUM_R + rbase;  // 8 contiguous floats
    *(float4*)po = o0;
    *((float4*)(po + 4)) = o1;
}

// ---------------------------------------------------------------------------
extern "C" void kernel_launch(void* const* d_in, const int* in_sizes, int n_in,
                              void* d_out, int out_size, void* d_ws, size_t ws_size,
                              hipStream_t stream) {
    const float* rts  = (const float*)d_in[0];   // [100000, 64]
    const float* tls  = (const float*)d_in[1];   // [64, 100000]
    const float* bias = (const float*)d_in[2];   // [100000]
    const int*   ridx = (const int*)d_in[3];     // [8192]
    const int*   lidx = (const int*)d_in[4];     // [8192]
    float* out = (float*)d_out;                  // [8192, 8192] (L, R)

    char* ws = (char*)d_ws;
    float*    lse  = (float*)(ws);                       // 64 * 4    = 256 B
    float*    ma   = (float*)(ws + 256);                 // 8192 * 4  = 32 KB
    float*    mb   = (float*)(ws + 256 + 32768);         // 8192 * 4  = 32 KB
    _Float16* expA = (_Float16*)(ws + 65792);            // 8192*64*2 = 1 MB
    _Float16* expB = (_Float16*)(ws + 65792 + 1048576);  // 8192*64*2 = 1 MB

    // 1) per-type logsumexp over full vocab
    lse_kernel<<<NUM_K, 256, 0, stream>>>(tls, bias, lse);

    // 2) A-side prep: 8192 rows, one wave each -> 1024 blocks of 256
    prep_a_kernel<<<NUM_R * 32 / 256, 256, 0, stream>>>(rts, ridx, expA, ma);

    // 3) B-side prep: 8192 cols, one wave each
    prep_b_kernel<<<NUM_L * 32 / 256, 256, 0, stream>>>(tls, bias, lidx, lse, expB, mb);

    // 4) WMMA GEMM + log epilogue: 512*512 tiles, 8 waves per block
    gemm_log_kernel<<<(512 * 512) / 8, 256, 0, stream>>>(expA, expB, ma, mb, out);
}
